// Self_Attn_71485435675250
// MI455X (gfx1250) — compile-verified, matrix operand order checked by reading, not run on hardware
//
#include <hip/hip_runtime.h>
#include <hip/hip_bf16.h>

typedef _Float16 v16h __attribute__((ext_vector_type(16)));
typedef _Float16 v8h  __attribute__((ext_vector_type(8)));
typedef float    v8f  __attribute__((ext_vector_type(8)));
typedef unsigned int u32x4 __attribute__((ext_vector_type(4)));
typedef int          i32x8 __attribute__((ext_vector_type(8)));
typedef int          i32x4 __attribute__((ext_vector_type(4)));

#define NSEQ 4096   // 64*64 spatial after downsample
#define CCH  64     // channels

// ---------------------------------------------------------------------------
// A/B fragment loader for V_WMMA_F32_16X16X32_F16.
// 16-bit A-matrix 16x32 layout (ISA 7.12.2): lane L holds row (L&15);
// elems 0..7 -> K = kb..kb+7, elems 8..15 -> K = kb+16..kb+23, kb = 8*(L>=16).
// B fragment uses the symmetric layout with lane = column index.
// Works on global or LDS pointers (addrspace inferred -> b128 loads).
// ---------------------------------------------------------------------------
__device__ __forceinline__ v16h load_frag(const _Float16* base, int ldm, int kfrag) {
  const int lane = threadIdx.x & 31;
  const int row  = lane & 15;
  const int kb   = kfrag * 32 + ((lane >> 4) << 3);
  const _Float16* p = base + row * ldm + kb;
  v16h r;
  ((v8h*)&r)[0] = *(const v8h*)(p);        // K = kb .. kb+7
  ((v8h*)&r)[1] = *(const v8h*)(p + 16);   // K = kb+16 .. kb+23
  return r;
}

__device__ __forceinline__ v8f wmma_f16(v16h a, v16h b, v8f c) {
  return __builtin_amdgcn_wmma_f32_16x16x32_f16(false, a, false, b,
                                                (short)0, c, false, false);
}

// ---------------------------------------------------------------------------
// Tensor Data Mover: DMA one 64x64 f16 tile (row stride `stride_elems`) from
// global memory into LDS at byte offset `lds_off`. D# per ISA ch.8:
//  group0: [1:0]=count=1, [63:32]=lds_addr, [120:64]=global_addr, [127:126]=2
//  group1: data_size=2B, tensor_dim0/1=64, tile_dim0/1=64,
//          tensor_dim0_stride=stride_elems; trailing groups zero (<=2D tensor).
// Tracked on TENSORcnt; EXEC ignored (issue from one wave).
// NOTE: driver toolchain exposes the 6-arg builtin (clang-23 / therock-10.0).
// ---------------------------------------------------------------------------
__device__ __forceinline__ void tdm_tile64(const _Float16* g, unsigned lds_off,
                                           int stride_elems) {
  unsigned long long ga = (unsigned long long)g;
  u32x4 g0 = { 1u, lds_off, (unsigned)ga,
               (unsigned)(ga >> 32) | (2u << 30) };
  i32x8 g1 = { (int)(1u << 16),     // data_size = 2 bytes
               (int)(64u << 16),    // tensor_dim0 = 64 (low16 in [63:48])
               (int)(64u << 16),    // tensor_dim1 = 64 (low16 in [95:80])
               (int)(64u << 16),    // tile_dim0  = 64 ([127:112])
               64,                  // tile_dim1  = 64 ([143:128])
               stride_elems,        // tensor_dim0_stride low32 ([191:160])
               0, 0 };
  i32x4 z4 = { 0, 0, 0, 0 };
  i32x8 z8 = { 0, 0, 0, 0, 0, 0, 0, 0 };
  __builtin_amdgcn_tensor_load_to_lds(g0, g1, z4, z4, z8, 0);
}

// ---------------------------------------------------------------------------
// Kernel 1: depthwise 3x3 stride-2 pad-1 conv, 128x128 -> 64x64, f32.
// ---------------------------------------------------------------------------
__global__ void __launch_bounds__(256) k_down(const float* __restrict__ x,
                                              const float* __restrict__ w,
                                              const float* __restrict__ bias,
                                              float* __restrict__ xd) {
  int idx = blockIdx.x * 256 + threadIdx.x;
  if (idx >= 4 * CCH * 64 * 64) return;
  int ox = idx & 63, oy = (idx >> 6) & 63, c = (idx >> 12) & 63, b = idx >> 18;
  const float* xin = x + ((b * CCH + c) * 128) * 128;
  const float* wc  = w + c * 9;
  float acc = bias[c];
  int iy0 = oy * 2 - 1, ix0 = ox * 2 - 1;
#pragma unroll
  for (int ky = 0; ky < 3; ky++) {
    int iy = iy0 + ky;
    if ((unsigned)iy >= 128u) continue;
#pragma unroll
    for (int kx = 0; kx < 3; kx++) {
      int ix = ix0 + kx;
      if ((unsigned)ix >= 128u) continue;
      acc += xin[iy * 128 + ix] * wc[ky * 3 + kx];
    }
  }
  xd[idx] = acc;
}

// ---------------------------------------------------------------------------
// Kernel 2: 1x1-conv projections. Q,K as [b][n][c] f16 ("N x d" row-major),
// V as [b][c][n] f16 ("d x N" row-major).
// ---------------------------------------------------------------------------
__global__ void __launch_bounds__(256) k_proj(
    const float* __restrict__ xd,
    const float* __restrict__ qw, const float* __restrict__ qb,
    const float* __restrict__ kw, const float* __restrict__ kb,
    const float* __restrict__ vw, const float* __restrict__ vb,
    _Float16* __restrict__ Qh, _Float16* __restrict__ Kh,
    _Float16* __restrict__ Vh) {
  __shared__ float tile[CCH][65];
  const int b  = blockIdx.x >> 6;
  const int n0 = (blockIdx.x & 63) * 64;
  const int tid = threadIdx.x;
  for (int t = tid; t < CCH * 64; t += 256) {
    int ic = t >> 6, ns = t & 63;
    tile[ic][ns] = xd[(b * CCH + ic) * NSEQ + n0 + ns];
  }
  __syncthreads();
  const int c = tid & 63;
  const int nbase = (tid >> 6) * 16;
  float aq[16], ak[16], av[16];
#pragma unroll
  for (int k = 0; k < 16; k++) { aq[k] = qb[c]; ak[k] = kb[c]; av[k] = vb[c]; }
  for (int ic = 0; ic < CCH; ic++) {
    float wq = qw[c * CCH + ic], wk = kw[c * CCH + ic], wv = vw[c * CCH + ic];
#pragma unroll
    for (int k = 0; k < 16; k++) {
      float xv = tile[ic][nbase + k];
      aq[k] += wq * xv; ak[k] += wk * xv; av[k] += wv * xv;
    }
  }
#pragma unroll
  for (int k = 0; k < 16; k++) {
    int n = n0 + nbase + k;
    Qh[(b * NSEQ + n) * CCH + c] = (_Float16)aq[k];
    Kh[(b * NSEQ + n) * CCH + c] = (_Float16)ak[k];
    Vh[(b * CCH + c) * NSEQ + n] = (_Float16)av[k];
  }
}

// ---------------------------------------------------------------------------
// Kernel 3: fused flash-attention. 8 waves/block, one 16-query tile per wave;
// all 8 waves share the same batch, so K/V 64x64 f16 tiles are staged ONCE
// per block by the Tensor Data Mover into double-buffered LDS (32 KB) while
// waves compute from the other buffer (TENSORcnt overlap).
// S^T = K_j*Q_i^T (j on the VGPR axis) -> per-lane softmax + 1 shfl_xor(16),
// and P^T B-fragments repack per-lane from D-layout (no transpose).
// ---------------------------------------------------------------------------
__global__ void __launch_bounds__(256, 1) k_attn(
    const _Float16* __restrict__ Qh, const _Float16* __restrict__ Kh,
    const _Float16* __restrict__ Vh, float* __restrict__ attw) {
  __shared__ _Float16 smem[2][2][64 * 64];   // [buf][K/V][tile] = 32 KB
  const int wave = threadIdx.x >> 5;
  const int g    = blockIdx.x * 8 + wave;    // 1024 waves total
  const int b    = g >> 8;                   // 256 i-tiles per batch; 8 per
  const int i0   = (g & 255) * 16;           // block -> single b per block
  const int lane = threadIdx.x & 31;

  const _Float16* Qb = Qh + (b * NSEQ) * CCH;
  const _Float16* Kb = Kh + (b * NSEQ) * CCH;
  const _Float16* Vb = Vh + (b * CCH) * NSEQ;

  // Q as B-fragments (lane = query column, K-dim = channel), held for all j.
  const v16h qb0 = load_frag(Qb + i0 * CCH, CCH, 0);
  const v16h qb1 = load_frag(Qb + i0 * CCH, CCH, 1);

  v8f o0 = {}, o1 = {}, o2 = {}, o3 = {};    // O^T: c-rows 0..63 x 16 i-cols
  float m_run = -1e30f, l_run = 0.f;         // per i-column (lane&15) stats

  const int NSTEP = NSEQ / 64;
  if (wave == 0) {                           // prologue DMA into buffer 0
    tdm_tile64(Kb, (unsigned)(unsigned long long)&smem[0][0][0], CCH);
    tdm_tile64(Vb, (unsigned)(unsigned long long)&smem[0][1][0], NSEQ);
  }

  for (int step = 0; step < NSTEP; ++step) {
    const int cur = step & 1;
    if (wave == 0) __builtin_amdgcn_s_wait_tensorcnt(0);  // buf[cur] landed
    __syncthreads();  // all waves: buf[cur] valid AND buf[cur^1] reads done
    if (wave == 0 && step + 1 < NSTEP) {     // DMA next tiles into buf[cur^1]
      const int j1 = (step + 1) * 64;
      tdm_tile64(Kb + j1 * CCH, (unsigned)(unsigned long long)&smem[cur ^ 1][0][0], CCH);
      tdm_tile64(Vb + j1,       (unsigned)(unsigned long long)&smem[cur ^ 1][1][0], NSEQ);
    }
    const _Float16* Kt = &smem[cur][0][0];   // [j-local 64][c 64]
    const _Float16* Vt = &smem[cur][1][0];   // [c 64][j-local 64]

    // S^T tile: 4 fragments, fragment f covers j-rows [16f, 16f+16)
    v8f st0, st1, st2, st3;
    {
      v8f z = {};
      v16h a0 = load_frag(Kt +  0 * CCH, CCH, 0);
      v16h a1 = load_frag(Kt +  0 * CCH, CCH, 1);
      st0 = wmma_f16(a1, qb1, wmma_f16(a0, qb0, z));
      a0 = load_frag(Kt + 16 * CCH, CCH, 0);
      a1 = load_frag(Kt + 16 * CCH, CCH, 1);
      st1 = wmma_f16(a1, qb1, wmma_f16(a0, qb0, z));
      a0 = load_frag(Kt + 32 * CCH, CCH, 0);
      a1 = load_frag(Kt + 32 * CCH, CCH, 1);
      st2 = wmma_f16(a1, qb1, wmma_f16(a0, qb0, z));
      a0 = load_frag(Kt + 48 * CCH, CCH, 0);
      a1 = load_frag(Kt + 48 * CCH, CCH, 1);
      st3 = wmma_f16(a1, qb1, wmma_f16(a0, qb0, z));
    }
    // Online softmax along j (VGPR axis + fragment axis + lane-group pair).
    float mloc = -1e30f;
#pragma unroll
    for (int r = 0; r < 8; r++)
      mloc = fmaxf(mloc, fmaxf(fmaxf(st0[r], st1[r]), fmaxf(st2[r], st3[r])));
    mloc = fmaxf(mloc, __shfl_xor(mloc, 16, 32));
    float m_new = fmaxf(m_run, mloc);
    float sc = __expf(m_run - m_new);
    float lloc = 0.f;
#pragma unroll
    for (int r = 0; r < 8; r++) {
      st0[r] = __expf(st0[r] - m_new);
      st1[r] = __expf(st1[r] - m_new);
      st2[r] = __expf(st2[r] - m_new);
      st3[r] = __expf(st3[r] - m_new);
      lloc += st0[r] + st1[r] + st2[r] + st3[r];
    }
    lloc += __shfl_xor(lloc, 16, 32);
    l_run = l_run * sc + lloc;
    m_run = m_new;
#pragma unroll
    for (int r = 0; r < 8; r++) { o0[r] *= sc; o1[r] *= sc; o2[r] *= sc; o3[r] *= sc; }

    // Per-lane repack: D-layout S^T fragments -> B-fragments of P^T
    // (D row r+8g <-> B K-elem e+8g: identical lane-group structure).
    v16h pb0, pb1;
#pragma unroll
    for (int e = 0; e < 8; e++) {
      pb0[e]     = (_Float16)st0[e];
      pb0[e + 8] = (_Float16)st1[e];
      pb1[e]     = (_Float16)st2[e];
      pb1[e + 8] = (_Float16)st3[e];
    }
    // O^T += V (A-frags, rows = channels, K = j-local) * P^T (B-frags)
    {
      v16h va0 = load_frag(Vt +  0 * 64, 64, 0);
      v16h va1 = load_frag(Vt +  0 * 64, 64, 1);
      o0 = wmma_f16(va1, pb1, wmma_f16(va0, pb0, o0));
      va0 = load_frag(Vt + 16 * 64, 64, 0);
      va1 = load_frag(Vt + 16 * 64, 64, 1);
      o1 = wmma_f16(va1, pb1, wmma_f16(va0, pb0, o1));
      va0 = load_frag(Vt + 32 * 64, 64, 0);
      va1 = load_frag(Vt + 32 * 64, 64, 1);
      o2 = wmma_f16(va1, pb1, wmma_f16(va0, pb0, o2));
      va0 = load_frag(Vt + 48 * 64, 64, 0);
      va1 = load_frag(Vt + 48 * 64, 64, 1);
      o3 = wmma_f16(va1, pb1, wmma_f16(va0, pb0, o3));
    }
  }
  // Normalize and scatter O^T (D-layout: c = 16f + r + 8*(lane>=16), i = lane&15)
  const float inv = 1.0f / l_run;
  float* ob = attw + (b * CCH) * NSEQ;
  const int icol = i0 + (lane & 15);
  const int cgrp = (lane >> 4) * 8;
#pragma unroll
  for (int r = 0; r < 8; r++) {
    ob[( 0 + cgrp + r) * NSEQ + icol] = o0[r] * inv;
    ob[(16 + cgrp + r) * NSEQ + icol] = o1[r] * inv;
    ob[(32 + cgrp + r) * NSEQ + icol] = o2[r] * inv;
    ob[(48 + cgrp + r) * NSEQ + icol] = o3[r] * inv;
  }
}

// ---------------------------------------------------------------------------
// Kernel 4: depthwise ConvTranspose2d(k=4, s=2, p=1) + bias, gamma*out + x.
// Gather form: out[oy,ox] += in[iy,ix]*w[ky,kx] with oy = 2*iy + ky - 1.
// ---------------------------------------------------------------------------
__global__ void __launch_bounds__(256) k_up(const float* __restrict__ attw,
                                            const float* __restrict__ dw,
                                            const float* __restrict__ db,
                                            const float* __restrict__ gamma,
                                            const float* __restrict__ in_x,
                                            float* __restrict__ out) {
  int idx = blockIdx.x * 256 + threadIdx.x;
  if (idx >= 4 * CCH * 128 * 128) return;
  int ox = idx & 127, oy = (idx >> 7) & 127, c = (idx >> 14) & 63, b = idx >> 20;
  const float* a  = attw + (b * CCH + c) * NSEQ;
  const float* wc = dw + c * 16;
  float acc = 0.f;
#pragma unroll
  for (int ky = 0; ky < 4; ky++) {
    int ty = oy + 1 - ky;
    if (ty & 1) continue;
    int iy = ty >> 1;
    if ((unsigned)iy >= 64u) continue;
#pragma unroll
    for (int kx = 0; kx < 4; kx++) {
      int tx = ox + 1 - kx;
      if (tx & 1) continue;
      int ix = tx >> 1;
      if ((unsigned)ix >= 64u) continue;
      acc += a[iy * 64 + ix] * wc[ky * 4 + kx];
    }
  }
  out[idx] = gamma[0] * (acc + db[c]) + in_x[idx];
}

// ---------------------------------------------------------------------------
extern "C" void kernel_launch(void* const* d_in, const int* in_sizes, int n_in,
                              void* d_out, int out_size, void* d_ws, size_t ws_size,
                              hipStream_t stream) {
  const float* in_x  = (const float*)d_in[0];
  const float* gw    = (const float*)d_in[1];
  const float* gb    = (const float*)d_in[2];
  const float* qw    = (const float*)d_in[3];
  const float* qbias = (const float*)d_in[4];
  const float* kw    = (const float*)d_in[5];
  const float* kbias = (const float*)d_in[6];
  const float* vw    = (const float*)d_in[7];
  const float* vbias = (const float*)d_in[8];
  const float* dw    = (const float*)d_in[9];
  const float* db    = (const float*)d_in[10];
  const float* gamma = (const float*)d_in[11];
  float* out = (float*)d_out;

  // Workspace layout (~14 MB total):
  char* ws = (char*)d_ws;
  float*    xd   = (float*)(ws);                         // 4*64*4096 f32 = 4 MB
  _Float16* Qh   = (_Float16*)(ws + 4u  * 1024 * 1024);  // 2 MB  [b][n][c]
  _Float16* Kh   = (_Float16*)(ws + 6u  * 1024 * 1024);  // 2 MB  [b][n][c]
  _Float16* Vh   = (_Float16*)(ws + 8u  * 1024 * 1024);  // 2 MB  [b][c][n]
  float*    attw = (float*)(ws + 10u * 1024 * 1024);     // 4 MB  [b][c][n]

  k_down<<<4096, 256, 0, stream>>>(in_x, gw, gb, xd);
  k_proj<<<256, 256, 0, stream>>>(xd, qw, qbias, kw, kbias, vw, vbias, Qh, Kh, Vh);
  k_attn<<<128, 256, 0, stream>>>(Qh, Kh, Vh, attw);   // 1024 waves, TDM-staged K/V
  k_up<<<16384, 256, 0, stream>>>(attw, dw, db, gamma, in_x, out);
}
